// GCN_DGG_35820027248990
// MI455X (gfx1250) — matrix operand order repaired and checked by reading, not compiled, not run on hardware
//
#include <hip/hip_runtime.h>
#include <hip/hip_bf16.h>

// GCN on MI455X (gfx1250): h1 = relu(D^-.5 A_hat D^-.5 X W1), h2 = relu(D^-.5 A_hat D^-.5 h1 W2),
// out = log_softmax(h2). Reassociated so the big SpMM is A_hat @ (scaled feature panel),
// streamed from HBM (memory-bound, 23.3 TB/s) and fed to v_wmma_f32_16x16x32_bf16.

#define GN     12288
#define NFEAT  128
#define NHID   64
#define NCLASS 16
#define KSPLIT 4

typedef __attribute__((ext_vector_type(16))) __bf16 v16bf;
typedef __attribute__((ext_vector_type(8)))  float  v8f;

// ---------------------------------------------------------------------------
// Fragment loaders (wave32 WMMA; A and B both use contiguous k = 16*(lane/16) + e)
// ---------------------------------------------------------------------------

// A fragment 16x32 bf16 from f32 row-major source (ld = row stride in floats).
// lanes 0-15 : M=lane,    K in {kb+0..15}
// lanes 16-31: M=lane-16, K in {kb+16..31}
// If grow >= 0, inject the +1 identity term when global column k == grow.
static __device__ inline v16bf load_a_frag_f32(const float* __restrict__ rowbase,
                                               int ld, int kb, int lane, int grow) {
  const int r    = lane & 15;
  const int koff = (lane < 16) ? 0 : 16;
  const float* p = rowbase + (size_t)r * ld + kb + koff;
  float4 f0 = *(const float4*)(p + 0);
  float4 f1 = *(const float4*)(p + 4);
  float4 f2 = *(const float4*)(p + 8);
  float4 f3 = *(const float4*)(p + 12);
  float v[16] = {f0.x, f0.y, f0.z, f0.w, f1.x, f1.y, f1.z, f1.w,
                 f2.x, f2.y, f2.z, f2.w, f3.x, f3.y, f3.z, f3.w};
  if (grow >= 0) {
    const int k0 = kb + koff;
#pragma unroll
    for (int e = 0; e < 16; ++e) {
      const int k = k0 + e;
      if (k == grow) v[e] += 1.0f;
    }
  }
  v16bf a;
#pragma unroll
  for (int e = 0; e < 16; ++e) a[e] = (__bf16)v[e];
  return a;
}

// B fragment 32x16 bf16 from transposed bf16 panel Bt[NW][GN] (row = n, contiguous k).
// lanes 0-15 : N=lane,    K=kb+0..15 ; lanes 16-31: N=lane-16, K=kb+16..31
static __device__ inline v16bf load_b_frag_bf16(const __bf16* __restrict__ Bt,
                                                int kb, int t, int lane) {
  const int n  = (t << 4) + (lane & 15);
  const int ks = kb + ((lane < 16) ? 0 : 16);
  const __bf16* p = Bt + (size_t)n * GN + ks;   // 32 contiguous bytes per lane
  union { uint4 u[2]; v16bf v; } c;
  c.u[0] = *(const uint4*)(p);
  c.u[1] = *(const uint4*)(p + 8);
  return c.v;
}

// B fragment 32x16 bf16 from small f32 row-major weight W[K][nw] (L2-resident).
static __device__ inline v16bf load_b_frag_w(const float* __restrict__ W, int nw,
                                             int kb, int t, int lane) {
  const int n  = (t << 4) + (lane & 15);
  const int ks = kb + ((lane < 16) ? 0 : 16);
  v16bf b;
#pragma unroll
  for (int e = 0; e < 16; ++e) b[e] = (__bf16)W[(size_t)(ks + e) * nw + n];
  return b;
}

// ---------------------------------------------------------------------------
// Kernel 1: d[i] = rsqrt(1 + rowsum(adj_i))   (A_hat = adj + I, diag(adj)=0)
// ---------------------------------------------------------------------------
__global__ void __launch_bounds__(256) degree_kernel(const float* __restrict__ adj,
                                                     float* __restrict__ dsc) {
  const int row = blockIdx.x;
  const float4* rp = (const float4*)(adj + (size_t)row * GN);
  float s = 0.f;
  for (int i = threadIdx.x; i < GN / 4; i += 256) {
    float4 v = rp[i];
    s += v.x + v.y + v.z + v.w;
  }
  __shared__ float red[256];
  red[threadIdx.x] = s;
  __syncthreads();
  for (int off = 128; off > 0; off >>= 1) {
    if (threadIdx.x < off) red[threadIdx.x] += red[threadIdx.x + off];
    __syncthreads();
  }
  if (threadIdx.x == 0) dsc[row] = rsqrtf(1.0f + red[0]);
}

// ---------------------------------------------------------------------------
// Kernel 2/5: YsT[n][m] = d[m] * (A @ W)[m][n]  (WMMA, bf16 in / f32 acc)
// A: [GN, KDIM] f32, W: [KDIM, NT*16] f32, YsT: [NT*16, GN] bf16 (transposed)
// One wave per 16-row M tile; 8 waves / block; grid.x = GN/16/8 = 96.
// ---------------------------------------------------------------------------
template <int KDIM, int NT>
__global__ void __launch_bounds__(256)
xw_scale_kernel(const float* __restrict__ A, const float* __restrict__ W,
                const float* __restrict__ dsc, __bf16* __restrict__ YsT) {
  const int lane = threadIdx.x & 31;
  const int wid  = threadIdx.x >> 5;
  const int m0   = (blockIdx.x * 8 + wid) * 16;
  v8f acc[NT];
#pragma unroll
  for (int t = 0; t < NT; ++t) acc[t] = (v8f){0, 0, 0, 0, 0, 0, 0, 0};
  const float* rowbase = A + (size_t)m0 * KDIM;
#pragma unroll
  for (int kb = 0; kb < KDIM; kb += 32) {
    v16bf a = load_a_frag_f32(rowbase, KDIM, kb, lane, -1);
#pragma unroll
    for (int t = 0; t < NT; ++t) {
      v16bf b = load_b_frag_w(W, NT * 16, kb, t, lane);
      acc[t] = __builtin_amdgcn_wmma_f32_16x16x32_bf16(
          false, a, false, b, (short)0, acc[t], false, false);
    }
  }
  // D layout: lane n = lane&15, VGPR r holds M = r + (lane<16 ? 0 : 8)
  const int nl   = lane & 15;
  const int moff = (lane < 16) ? 0 : 8;
  float4 d0 = *(const float4*)(dsc + m0 + moff);
  float4 d1 = *(const float4*)(dsc + m0 + moff + 4);
  float dv[8] = {d0.x, d0.y, d0.z, d0.w, d1.x, d1.y, d1.z, d1.w};
#pragma unroll
  for (int t = 0; t < NT; ++t) {
    union { __bf16 h[8]; uint4 u; } pk;
#pragma unroll
    for (int r = 0; r < 8; ++r) pk.h[r] = (__bf16)(acc[t][r] * dv[r]);
    *(uint4*)(YsT + (size_t)((t << 4) + nl) * GN + m0 + moff) = pk.u;
  }
}

// ---------------------------------------------------------------------------
// Kernel 3/6: partial[split][m][n] = (A_hat @ Ys)[m][n] over K chunk `split`.
// Streams 604 MB adj f32 from HBM, converts to bf16 in-register, WMMA bf16.
// One wave = 16 rows x (NT*16) cols; grid = (96, KSPLIT).
// ---------------------------------------------------------------------------
template <int NT>
__global__ void __launch_bounds__(256)
spmm_kernel(const float* __restrict__ adj, const __bf16* __restrict__ Bt,
            float* __restrict__ part) {
  const int lane  = threadIdx.x & 31;
  const int wid   = threadIdx.x >> 5;
  const int m0    = (blockIdx.x * 8 + wid) * 16;
  const int split = blockIdx.y;
  const int kbeg  = split * (GN / KSPLIT);
  const int kend  = kbeg + (GN / KSPLIT);
  const int grow  = m0 + (lane & 15);

  v8f acc[NT];
#pragma unroll
  for (int t = 0; t < NT; ++t) acc[t] = (v8f){0, 0, 0, 0, 0, 0, 0, 0};

  const float* rowbase = adj + (size_t)m0 * GN;
  const int koff = (lane < 16) ? 0 : 8;
#pragma unroll 2
  for (int kb = kbeg; kb < kend; kb += 32) {
    // stream-ahead prefetch of this lane's adj row (emits global_prefetch_b8)
    __builtin_prefetch(
        (const void*)(rowbase + (size_t)(lane & 15) * GN + kb + koff + 64), 0, 0);
    v16bf a = load_a_frag_f32(rowbase, GN, kb, lane, grow);
#pragma unroll
    for (int t = 0; t < NT; ++t) {
      v16bf b = load_b_frag_bf16(Bt, kb, t, lane);
      acc[t] = __builtin_amdgcn_wmma_f32_16x16x32_bf16(
          false, a, false, b, (short)0, acc[t], false, false);
    }
  }

  const int nl   = lane & 15;
  const int moff = (lane < 16) ? 0 : 8;
  float* op = part + (size_t)split * GN * (NT * 16);
#pragma unroll
  for (int t = 0; t < NT; ++t)
#pragma unroll
    for (int r = 0; r < 8; ++r)
      op[(size_t)(m0 + moff + r) * (NT * 16) + (t << 4) + nl] = acc[t][r];
}

// ---------------------------------------------------------------------------
// Kernel 4: h1[j][h] = relu(d[j] * sum_splits Hpart)
// ---------------------------------------------------------------------------
__global__ void __launch_bounds__(256)
reduce_relu_kernel(const float* __restrict__ part, const float* __restrict__ dsc,
                   float* __restrict__ h1) {
  const int i = blockIdx.x * 256 + threadIdx.x;    // float4 index
  const size_t flat = (size_t)i * 4;
  const int j = (int)(flat >> 6);                  // / NHID (64)
  float4 s = {0, 0, 0, 0};
#pragma unroll
  for (int sp = 0; sp < KSPLIT; ++sp) {
    float4 v = *(const float4*)(part + (size_t)sp * GN * NHID + flat);
    s.x += v.x; s.y += v.y; s.z += v.z; s.w += v.w;
  }
  const float dj = dsc[j];
  float4 r = {fmaxf(s.x * dj, 0.f), fmaxf(s.y * dj, 0.f),
              fmaxf(s.z * dj, 0.f), fmaxf(s.w * dj, 0.f)};
  *(float4*)(h1 + flat) = r;
}

// ---------------------------------------------------------------------------
// Kernel 7: out[j] = log_softmax(relu(d[j] * sum_splits Gpart[j]))  (16 classes)
// 16-lane groups own one row; shfl_xor {1,2,4,8} reduces within the group.
// ---------------------------------------------------------------------------
__global__ void __launch_bounds__(256)
final_logsoftmax_kernel(const float* __restrict__ part, const float* __restrict__ dsc,
                        float* __restrict__ out) {
  const int lane = threadIdx.x & 31;
  const int wid  = threadIdx.x >> 5;
  const int j = blockIdx.x * 16 + wid * 2 + (lane >> 4);
  const int c = lane & 15;
  float v = 0.f;
#pragma unroll
  for (int sp = 0; sp < KSPLIT; ++sp)
    v += part[(size_t)sp * GN * NCLASS + (size_t)j * NCLASS + c];
  v = fmaxf(v * dsc[j], 0.f);
  float m = v;
#pragma unroll
  for (int o = 1; o < 16; o <<= 1) m = fmaxf(m, __shfl_xor(m, o, 32));
  const float e = __expf(v - m);
  float s = e;
#pragma unroll
  for (int o = 1; o < 16; o <<= 1) s += __shfl_xor(s, o, 32);
  out[(size_t)j * NCLASS + c] = (v - m) - __logf(s);
}

// ---------------------------------------------------------------------------
// Host launcher
// ---------------------------------------------------------------------------
extern "C" void kernel_launch(void* const* d_in, const int* in_sizes, int n_in,
                              void* d_out, int out_size, void* d_ws, size_t ws_size,
                              hipStream_t stream) {
  (void)in_sizes; (void)n_in; (void)out_size; (void)ws_size;
  const float* x   = (const float*)d_in[0];   // [GN, NFEAT]
  const float* adj = (const float*)d_in[1];   // [GN, GN]
  const float* W1  = (const float*)d_in[2];   // [NFEAT, NHID]
  const float* W2  = (const float*)d_in[3];   // [NHID, NCLASS]
  float* out = (float*)d_out;                 // [GN, NCLASS]

  // Workspace layout (all offsets 256B-aligned)
  char* ws = (char*)d_ws;
  size_t off = 0;
  float*  dsc   = (float*)(ws + off);  off += (size_t)GN * 4;                       // 48 KB
  __bf16* Y1sT  = (__bf16*)(ws + off); off += (size_t)NHID * GN * 2;                // 1.5 MB
  float*  Hpart = (float*)(ws + off);  off += (size_t)KSPLIT * GN * NHID * 4;       // 12 MB
  float*  h1    = (float*)(ws + off);  off += (size_t)GN * NHID * 4;                // 3 MB
  __bf16* Y2sT  = (__bf16*)(ws + off); off += (size_t)NCLASS * GN * 2;              // 384 KB
  float*  Gpart = (float*)(ws + off);  off += (size_t)KSPLIT * GN * NCLASS * 4;     // 3 MB

  // 1) degrees
  degree_kernel<<<GN, 256, 0, stream>>>(adj, dsc);
  // 2) Y1s^T = (d .* (x @ W1))^T  bf16
  xw_scale_kernel<NFEAT, NHID / 16><<<GN / 16 / 8, 256, 0, stream>>>(x, W1, dsc, Y1sT);
  // 3) Hpart = A_hat @ Y1s  (K-split partials)
  spmm_kernel<NHID / 16><<<dim3(GN / 16 / 8, KSPLIT), 256, 0, stream>>>(adj, Y1sT, Hpart);
  // 4) h1 = relu(d .* sum(Hpart))
  reduce_relu_kernel<<<(GN * NHID / 4) / 256, 256, 0, stream>>>(Hpart, dsc, h1);
  // 5) Y2s^T = (d .* (h1 @ W2))^T  bf16
  xw_scale_kernel<NHID, NCLASS / 16><<<GN / 16 / 8, 256, 0, stream>>>(h1, W2, dsc, Y2sT);
  // 6) Gpart = A_hat @ Y2s
  spmm_kernel<NCLASS / 16><<<dim3(GN / 16 / 8, KSPLIT), 256, 0, stream>>>(adj, Y2sT, Gpart);
  // 7) relu + log_softmax
  final_logsoftmax_kernel<<<GN / 16, 256, 0, stream>>>(Gpart, dsc, out);
}
